// LSTMModel_51024211476885
// MI455X (gfx1250) — matrix operand order repaired and checked by reading, not compiled, not run on hardware
//
#include <hip/hip_runtime.h>
#include <math.h>

// ---------------------------------------------------------------------------
// LSTM (T=1024,B=32,I=256,H=512) + Linear(O=2048) + log_softmax for gfx1250.
// All GEMMs via v_wmma_f32_16x16x32_f16 with fragment-pre-packed operands:
// each lane's 16 f16 elements are contiguous 32B records -> pure b128 traffic.
// Recurrent phase: persistent kernel, LDS-resident W_hh slice, async
// global->LDS h staging (ASYNCcnt path), cell state in registers.
// ---------------------------------------------------------------------------

typedef __attribute__((ext_vector_type(16))) _Float16 v16h;
typedef __attribute__((ext_vector_type(8)))  float    v8f;
typedef __attribute__((ext_vector_type(4)))  int      v4i;

#define T_STEPS 1024
#define BATCH   32
#define IN_DIM  256
#define HID     512
#define OUT_DIM 2048
#define G4H     2048          // 4*HID
#define REC_BLOCKS 16
#define HCOLS_PER_BLOCK 32    // HID / REC_BLOCKS
#define KC_H    (HID / 32)    // 16 k-chunks for K=512

#define AS1 __attribute__((address_space(1)))
#define AS3 __attribute__((address_space(3)))

#if defined(__AMDGCN__) && __has_builtin(__builtin_amdgcn_global_load_async_to_lds_b128)
#define ASYNC_OK 1
#else
#define ASYNC_OK 0
#endif

// ---------------------------------------------------------------- WMMA utils
__device__ __forceinline__ v8f wmma16x16x32(v16h a, v16h b, v8f c) {
  return __builtin_amdgcn_wmma_f32_16x16x32_f16(
      /*neg_a=*/false, a, /*neg_b=*/false, b,
      /*c_mod=*/(short)0, c, /*reuse_a=*/false, /*reuse_b=*/false);
}

// async 16B global->LDS copy (gfx1250 ASYNCcnt path) with sync fallback
__device__ __forceinline__ void cp_b128(void* lds, const void* g) {
#if ASYNC_OK
  __builtin_amdgcn_global_load_async_to_lds_b128((AS1 v4i*)g, (AS3 v4i*)lds,
                                                 0, 0);
#else
  *(uint4*)lds = *(const uint4*)g;
#endif
}
__device__ __forceinline__ void cp_wait() {
#if ASYNC_OK
#if __has_builtin(__builtin_amdgcn_s_wait_asynccnt)
  __builtin_amdgcn_s_wait_asynccnt(0);
#else
  asm volatile("s_wait_asynccnt 0x0" ::: "memory");
#endif
#endif
}

__device__ __forceinline__ float sigmoidf_(float x) {
  return 1.0f / (1.0f + __expf(-x));
}

// ----------------------------------------------------------- grid-wide sync
__device__ __forceinline__ void grid_sync(unsigned* ctr, int nblocks) {
  __syncthreads();
  if (threadIdx.x == 0) {
    __threadfence();
    unsigned arrived = atomicAdd(ctr, 1u) + 1u;
    if (arrived < (unsigned)nblocks) {
      while (__hip_atomic_load(ctr, __ATOMIC_RELAXED,
                               __HIP_MEMORY_SCOPE_AGENT) < (unsigned)nblocks)
        __builtin_amdgcn_s_sleep(1);
    }
    __threadfence();
  }
  __syncthreads();
}

// --------------------------------------------------- fragment packing kernels
// A-fragment record layout: Af[((m16*KC + kc)*32 + lane)*16 + e]
//   m = m16*16 + (lane&15); half = lane>>4
//   k = kc*32 + ((e>>3)*2 + half)*8 + (e&7)          (ISA 7.12.2 16-bit A)
__global__ __launch_bounds__(256) void pack_a_frag(
    const float* __restrict__ src, _Float16* __restrict__ dst, int M, int K) {
  const int KC = K >> 5;
  long long n = (long long)M * K;
  long long i = (long long)blockIdx.x * blockDim.x + threadIdx.x;
  long long stride = (long long)gridDim.x * blockDim.x;
  for (; i < n; i += stride) {
    int e = (int)(i & 15);
    long long rec = i >> 4;
    int lane = (int)(rec & 31);
    long long t = rec >> 5;
    int kc = (int)(t % KC);
    long long m16 = t / KC;
    int half = lane >> 4, mm = lane & 15;
    long long m = m16 * 16 + mm;
    int k = kc * 32 + ((e >> 3) * 2 + half) * 8 + (e & 7);
    dst[i] = (_Float16)src[m * K + k];
  }
}

// B-fragment record layout (B = W^T, src is W[N][K] row-major):
//   Bf[((n16*KC + kc)*32 + lane)*16 + e]
//   n = n16*16 + (lane&15); k = kc*32 + ((e>>3)*2 + (lane>>4))*8 + (e&7)
__global__ __launch_bounds__(256) void pack_b_frag(
    const float* __restrict__ src, _Float16* __restrict__ dst, int N, int K) {
  const int KC = K >> 5;
  long long n = (long long)N * K;
  long long i = (long long)blockIdx.x * blockDim.x + threadIdx.x;
  long long stride = (long long)gridDim.x * blockDim.x;
  for (; i < n; i += stride) {
    int e = (int)(i & 15);
    long long rec = i >> 4;
    int lane = (int)(rec & 31);
    long long t = rec >> 5;
    int kc = (int)(t % KC);
    long long n16 = t / KC;
    int nn = n16 * 16 + (lane & 15);
    int k = kc * 32 + ((e >> 3) * 2 + (lane >> 4)) * 8 + (e & 7);
    dst[i] = (_Float16)src[(size_t)nn * K + k];
  }
}

__global__ __launch_bounds__(256) void bias_sum_kernel(
    const float* __restrict__ a, const float* __restrict__ b,
    float* __restrict__ o, int n) {
  int i = blockIdx.x * blockDim.x + threadIdx.x;
  if (i < n) o[i] = a[i] + b[i];
}

// ------------------------------------------------- GEMM on packed fragments
// C[M][N] f32 = A*B + bias. Block: 256 thr / 8 waves, tile 64x128,
// wave tile 32x32 (2x2 WMMA). Operands stream from L2 as b128 pairs.
__global__ __launch_bounds__(256) void gemm_frag(
    const _Float16* __restrict__ Af, const _Float16* __restrict__ Bf,
    const float* __restrict__ bias, float* __restrict__ C, int N, int K) {
  const int KC = K >> 5;
  const int lane = threadIdx.x & 31;
  const int wave = threadIdx.x >> 5;
  const int wm = wave >> 2, wn = wave & 3;
  const int m16 = blockIdx.x * 4 + wm * 2;   // 64 rows per block
  const int n16 = blockIdx.y * 8 + wn * 2;   // 128 cols per block
  const _Float16* pa0 = Af + (((size_t)(m16 + 0) * KC) * 32 + lane) * 16;
  const _Float16* pa1 = Af + (((size_t)(m16 + 1) * KC) * 32 + lane) * 16;
  const _Float16* pb0 = Bf + (((size_t)(n16 + 0) * KC) * 32 + lane) * 16;
  const _Float16* pb1 = Bf + (((size_t)(n16 + 1) * KC) * 32 + lane) * 16;
  v8f acc[2][2] = {};
#pragma unroll 2
  for (int kc = 0; kc < KC; ++kc) {
    v16h a0 = *(const v16h*)(pa0 + (size_t)kc * 512);
    v16h a1 = *(const v16h*)(pa1 + (size_t)kc * 512);
    v16h b0 = *(const v16h*)(pb0 + (size_t)kc * 512);
    v16h b1 = *(const v16h*)(pb1 + (size_t)kc * 512);
    acc[0][0] = wmma16x16x32(a0, b0, acc[0][0]);
    acc[0][1] = wmma16x16x32(a0, b1, acc[0][1]);
    acc[1][0] = wmma16x16x32(a1, b0, acc[1][0]);
    acc[1][1] = wmma16x16x32(a1, b1, acc[1][1]);
  }
  const int half = lane >> 4, nn = lane & 15;
#pragma unroll
  for (int mt = 0; mt < 2; ++mt)
#pragma unroll
    for (int nt = 0; nt < 2; ++nt) {
      int row = (m16 + mt) * 16 + 8 * half;
      int col = (n16 + nt) * 16 + nn;
      float* dst = C + (size_t)row * N + col;
      float bv = bias[col];
#pragma unroll
      for (int e = 0; e < 8; ++e) dst[(size_t)e * N] = acc[mt][nt][e] + bv;
    }
}

// ------------------------------------------------------- persistent LSTM rec
// 16 blocks x 256 threads. Block owns 32 h-cols (128 gate cols). W_hh slice
// lives in LDS in B-fragment order for all 1024 steps; h double-buffered in
// global in A-fragment order (staging = contiguous 32KB async copy); cell
// state lives in registers; software grid barrier per step.
__global__ __launch_bounds__(256) void lstm_rec(
    const _Float16* __restrict__ whh_f,   // packed B-frag, N=2048,K=512
    const float*    __restrict__ xW,      // [T*B][G4H]
    _Float16*       __restrict__ hbuf,    // [2][A-frag 32x512]
    _Float16*       __restrict__ ysf,     // packed A-frag, M=32768,K=512
    const float*    __restrict__ h0,
    const float*    __restrict__ c0,
    float* __restrict__ hT_out, float* __restrict__ cT_out,
    unsigned* __restrict__ bars) {
  extern __shared__ char smem_raw[];
  _Float16* Ws = (_Float16*)smem_raw;                       // 16*8*32*16 halfs
  _Float16* Hs = (_Float16*)(smem_raw + KC_H * 8 * 32 * 16 * 2);  // 16*2*32*16
  float*    Gs = (float*)(smem_raw + KC_H * 8 * 32 * 16 * 2 + KC_H * 2 * 32 * 16 * 2);

  const int tid  = threadIdx.x;
  const int lane = tid & 31;
  const int wave = tid >> 5;
  const int c0col = blockIdx.x * HCOLS_PER_BLOCK;

  // Preload W_hh slice into LDS, fragment order: Ws[((kc*8+wave)*32+lane)*16]
  // Wave w covers local gate cols [w*16,w*16+16): gate g=w>>1, n16 global:
  const int n16g = (wave >> 1) * 32 + (c0col >> 4) + (wave & 1);
  for (int kc = 0; kc < KC_H; ++kc) {
    const _Float16* src = whh_f + (((size_t)n16g * KC_H + kc) * 32 + lane) * 16;
    _Float16* dst = Ws + (((kc * 8 + wave) * 32) + lane) * 16;
    cp_b128(dst, src);
    cp_b128(dst + 8, src + 8);
  }
  cp_wait();

  // Init cell/hidden registers + parity-0 h buffer (A-fragment order).
  float c_reg[4], h_reg[4];
#pragma unroll
  for (int p = 0; p < 4; ++p) {
    int flat = tid + p * 256;          // (b, jo)
    int b = flat >> 5, jo = flat & 31;
    int col = c0col + jo;
    c_reg[p] = c0[b * HID + col];
    float hv = h0[b * HID + col];
    h_reg[p] = hv;
    int kc = col >> 5, kk = col & 31;
    int hf = (kk >> 3) & 1;
    int e  = ((kk >> 4) << 3) + (kk & 7);
    int mt = b >> 4;
    int ln = (b & 15) + 16 * hf;
    hbuf[(((kc * 2) + mt) * 32 + ln) * 16 + e] = (_Float16)hv;
  }
  __threadfence();
  grid_sync(bars + 0, REC_BLOCKS);

  for (int t = 0; t < T_STEPS; ++t) {
    const int par = t & 1;
    const _Float16* hsrc = hbuf + par * (KC_H * 2 * 32 * 16);
    _Float16*       hdst = hbuf + (par ^ 1) * (KC_H * 2 * 32 * 16);

    // Stage h (A-frag, 32KB contiguous) into LDS.
#pragma unroll
    for (int q = 0; q < 8; ++q) {
      int u = (q * 256 + tid) * 8;          // 2048 x 16B units
      cp_b128(Hs + u, hsrc + u);
    }
    cp_wait();
    __syncthreads();

    // Gates GEMM: [32x128] = h[32x512] x Whh_slice[512x128]
    {
      v8f acc0 = {}, acc1 = {};
#pragma unroll 4
      for (int kc = 0; kc < KC_H; ++kc) {
        v16h bf = *(const v16h*)(Ws + (((kc * 8 + wave) * 32) + lane) * 16);
        v16h a0 = *(const v16h*)(Hs + (((kc * 2 + 0) * 32) + lane) * 16);
        v16h a1 = *(const v16h*)(Hs + (((kc * 2 + 1) * 32) + lane) * 16);
        acc0 = wmma16x16x32(a0, bf, acc0);
        acc1 = wmma16x16x32(a1, bf, acc1);
      }
      const int half = lane >> 4, nn = lane & 15;
      const int n0 = wave * 16;
#pragma unroll
      for (int e = 0; e < 8; ++e) {
        Gs[(e + 8 * half) * 128 + n0 + nn]      = acc0[e];
        Gs[(16 + e + 8 * half) * 128 + n0 + nn] = acc1[e];
      }
    }
    __syncthreads();

    // Fused LSTM cell; c in registers across all 1024 steps.
#pragma unroll
    for (int p = 0; p < 4; ++p) {
      int flat = tid + p * 256;
      int b = flat >> 5, jo = flat & 31;
      size_t xrow = ((size_t)t * BATCH + b) * G4H;
      float iv = Gs[b * 128 +      jo] + xW[xrow + 0 * HID + c0col + jo];
      float fv = Gs[b * 128 + 32 + jo] + xW[xrow + 1 * HID + c0col + jo];
      float gv = Gs[b * 128 + 64 + jo] + xW[xrow + 2 * HID + c0col + jo];
      float ov = Gs[b * 128 + 96 + jo] + xW[xrow + 3 * HID + c0col + jo];
      float cn = sigmoidf_(fv) * c_reg[p] + sigmoidf_(iv) * tanhf(gv);
      float hn = sigmoidf_(ov) * tanhf(cn);
      c_reg[p] = cn;
      h_reg[p] = hn;
      _Float16 hh = (_Float16)hn;
      int col = c0col + jo;
      int kc = col >> 5, kk = col & 31;
      int hf = (kk >> 3) & 1;
      int e  = ((kk >> 4) << 3) + (kk & 7);
      int mt = b >> 4;
      int ln = (b & 15) + 16 * hf;
      hdst[(((kc * 2) + mt) * 32 + ln) * 16 + e] = hh;      // next-step A-frag
      // ys in A-fragment order for phase-3 GEMM: m16 = 2t + (b>>4)
      ysf[(((size_t)(2 * t + mt) * KC_H + kc) * 32 + ln) * 16 + e] = hh;
    }
    __threadfence();
    grid_sync(bars + 1 + t, REC_BLOCKS);
  }

  // Final states straight from registers.
#pragma unroll
  for (int p = 0; p < 4; ++p) {
    int flat = tid + p * 256;
    int b = flat >> 5, jo = flat & 31;
    int col = c0col + jo;
    hT_out[b * HID + col] = h_reg[p];
    cT_out[b * HID + col] = c_reg[p];
  }
}

// ----------------------------------------------------------- log_softmax row
__global__ __launch_bounds__(256) void logsoftmax_rows(
    const float* __restrict__ logits, float* __restrict__ out) {
  __shared__ float red[256];
  const int row = blockIdx.x;
  const int tid = threadIdx.x;
  const float* src = logits + (size_t)row * OUT_DIM;
  float v[8];
  float m = -3.4e38f;
#pragma unroll
  for (int j = 0; j < 8; ++j) {
    v[j] = src[tid + j * 256];
    m = fmaxf(m, v[j]);
  }
  red[tid] = m;
  __syncthreads();
  for (int s = 128; s > 0; s >>= 1) {
    if (tid < s) red[tid] = fmaxf(red[tid], red[tid + s]);
    __syncthreads();
  }
  m = red[0];
  __syncthreads();
  float sum = 0.f;
#pragma unroll
  for (int j = 0; j < 8; ++j) sum += __expf(v[j] - m);
  red[tid] = sum;
  __syncthreads();
  for (int s = 128; s > 0; s >>= 1) {
    if (tid < s) red[tid] += red[tid + s];
    __syncthreads();
  }
  float lse = m + __logf(red[0]);
  float* dst = out + (size_t)row * OUT_DIM;
#pragma unroll
  for (int j = 0; j < 8; ++j) dst[tid + j * 256] = v[j] - lse;
}

// ----------------------------------------------------------------- launcher
extern "C" void kernel_launch(void* const* d_in, const int* in_sizes, int n_in,
                              void* d_out, int out_size, void* d_ws, size_t ws_size,
                              hipStream_t stream) {
  const float* x     = (const float*)d_in[0];
  const float* h0    = (const float*)d_in[1];
  const float* c0    = (const float*)d_in[2];
  const float* W_ih  = (const float*)d_in[3];
  const float* W_hh  = (const float*)d_in[4];
  const float* b_ih  = (const float*)d_in[5];
  const float* b_hh  = (const float*)d_in[6];
  const float* W_lin = (const float*)d_in[7];
  const float* b_lin = (const float*)d_in[8];
  float* out = (float*)d_out;

  char* ws = (char*)d_ws;
  size_t off = 0;
  auto carve = [&](size_t bytes) -> char* {
    char* p = ws + off;
    off = (off + bytes + 255) & ~(size_t)255;
    return p;
  };
  const size_t MN = (size_t)T_STEPS * BATCH;                          // 32768
  float*    xw_logits = (float*)   carve(MN * G4H * sizeof(float));   // reused
  _Float16* xf        = (_Float16*)carve(MN * IN_DIM * 2);            // A-frag
  _Float16* ysf       = (_Float16*)carve(MN * HID * 2);               // A-frag
  _Float16* wihf      = (_Float16*)carve((size_t)G4H * IN_DIM * 2);   // B-frag
  _Float16* whhf      = (_Float16*)carve((size_t)G4H * HID * 2);      // B-frag
  _Float16* wlinf     = (_Float16*)carve((size_t)OUT_DIM * HID * 2);  // B-frag
  float*    bsum      = (float*)   carve(G4H * sizeof(float));
  _Float16* hbuf      = (_Float16*)carve(2 * (size_t)BATCH * HID * 2);
  unsigned* bars      = (unsigned*)carve((T_STEPS + 1) * sizeof(unsigned));
  (void)ws_size; (void)in_sizes; (void)n_in; (void)out_size;

  float* hT_out = out + MN * OUT_DIM;
  float* cT_out = hT_out + (size_t)BATCH * HID;

  (void)hipMemsetAsync(bars, 0, (T_STEPS + 1) * sizeof(unsigned), stream);

  // One-time fragment packing (memory-bound, tiny vs 2x268MB outputs).
  pack_a_frag<<<4096, 256, 0, stream>>>(x, xf, (int)MN, IN_DIM);
  pack_b_frag<<<1024, 256, 0, stream>>>(W_ih,  wihf,  G4H, IN_DIM);
  pack_b_frag<<<2048, 256, 0, stream>>>(W_hh,  whhf,  G4H, HID);
  pack_b_frag<<<2048, 256, 0, stream>>>(W_lin, wlinf, OUT_DIM, HID);
  bias_sum_kernel<<<8, 256, 0, stream>>>(b_ih, b_hh, bsum, G4H);

  // Phase 1: xW = x @ W_ih^T + (b_ih + b_hh)
  gemm_frag<<<dim3(MN / 64, G4H / 128), 256, 0, stream>>>(
      xf, wihf, bsum, xw_logits, G4H, IN_DIM);

  // Phase 2: persistent recurrence.
  const size_t rec_smem = (size_t)KC_H * 8 * 32 * 16 * 2   // Whh slice 128KB
                        + (size_t)KC_H * 2 * 32 * 16 * 2   // staged h   32KB
                        + (size_t)BATCH * 128 * 4;         // gates      16KB
  lstm_rec<<<REC_BLOCKS, 256, rec_smem, stream>>>(
      whhf, xw_logits, hbuf, ysf, h0, c0, hT_out, cT_out, bars);

  // Phase 3: logits = ys @ W_lin^T + b_lin (reuse buffer), then log_softmax.
  gemm_frag<<<dim3(MN / 64, OUT_DIM / 128), 256, 0, stream>>>(
      ysf, wlinf, b_lin, xw_logits, OUT_DIM, HID);
  logsoftmax_rows<<<(int)MN, 256, 0, stream>>>(xw_logits, out);
}
